// SemiSeq_49194555408593
// MI455X (gfx1250) — compile-verified, matrix operand order checked by reading, not run on hardware
//
#include <hip/hip_runtime.h>

typedef __bf16 bf16;
typedef __attribute__((ext_vector_type(16))) __bf16 v16bf;
typedef __attribute__((ext_vector_type(8)))  float  v8f;

#define B_   64
#define T_   1024
#define DIN_ 64
#define H_   256
#define G_   768       // 3*H (gates)
#define HH_  512       // 2*H (bidir concat)
#define M_   (B_ * T_)
#define WROWS_ 448     // Whh rows staged in LDS (gate r + 3/4 of z)
#define LP_    264     // padded LDS row pitch (elements): bank-conflict-free

union Frag16 { v16bf v; unsigned int u[8]; unsigned short s[16]; };

__device__ __forceinline__ unsigned short f2bf_bits(float f) {
  unsigned int u = __builtin_bit_cast(unsigned int, f);
  u += 0x7fffu + ((u >> 16) & 1u);          // round-to-nearest-even
  return (unsigned short)(u >> 16);
}
__device__ __forceinline__ float bf2f(unsigned short h) {
  unsigned int u = ((unsigned int)h) << 16;
  return __builtin_bit_cast(float, u);
}
__device__ __forceinline__ float sigmoid_fast(float x) {
  return __builtin_amdgcn_rcpf(1.f + __expf(-x));
}
__device__ __forceinline__ float tanh_fast(float x) {
  const float e = __expf(2.f * x);
  return 1.f - 2.f * __builtin_amdgcn_rcpf(e + 1.f);
}

// CDNA5 async global->LDS copy (tracked by ASYNCcnt).
__device__ __forceinline__ void async_ld_lds_b128(unsigned lds_addr,
                                                  const void* gaddr) {
  unsigned long long ga = (unsigned long long)(size_t)gaddr;
  asm volatile("global_load_async_to_lds_b128 %0, %1, off"
               :: "v"(lds_addr), "v"(ga) : "memory");
}
__device__ __forceinline__ void wait_async0() {
  asm volatile("s_wait_asynccnt 0x0" ::: "memory");
}

// 16x32 bf16 WMMA fragment load. The "major" index (row of A, column of B,
// since B is consumed K-major) is owned per-lane; K pairing follows the
// CDNA5 16-bit operand layout: vgpr pair p holds
//   K = (p&3)*2 + (p>=4 ? 16 : 0) + (lane>=16 ? 8 : 0)  (and K+1)
__device__ __forceinline__ v16bf load_frag_bf16(const bf16* __restrict__ base,
                                                int ld, int lane) {
  Frag16 f;
  const bf16* row = base + (size_t)(lane & 15) * ld + ((lane >> 4) << 3);
#pragma unroll
  for (int p = 0; p < 8; ++p) {
    const int k = ((p & 3) << 1) + ((p >> 2) << 4);
    f.u[p] = *(const unsigned int*)(row + k);   // two adjacent bf16 per dword
  }
  return f.v;
}

__device__ __forceinline__ v8f wmma_bf16(v16bf a, v16bf b, v8f c) {
  return __builtin_amdgcn_wmma_f32_16x16x32_bf16(false, a, false, b, (short)0, c,
                                                 false, false);
}

// ---------------------------------------------------------------------------
// f32 -> bf16 elementwise convert
// ---------------------------------------------------------------------------
__global__ __launch_bounds__(256) void cvt_bf16(const float* __restrict__ in,
                                                bf16* __restrict__ out, int n) {
  int i = blockIdx.x * 256 + threadIdx.x;
  if (i < n) out[i] = __builtin_bit_cast(bf16, f2bf_bits(in[i]));
}

// ---------------------------------------------------------------------------
// Input projection GEMM: O[M_ x G_] = X[M_ x K] * W[G_ x K]^T  (f32 out)
// 8 waves/block as 2(M) x 4(N); each wave owns a 32x64 tile (2x4 WMMA tiles).
// Block tile 64 x 256; grid (M_/64, G_/256).
// ---------------------------------------------------------------------------
__global__ __launch_bounds__(256) void xproj_gemm(const bf16* __restrict__ X,
                                                  const bf16* __restrict__ W,
                                                  float* __restrict__ O, int K) {
  const int lane = threadIdx.x & 31;
  const int wv   = threadIdx.x >> 5;
  const int m0 = blockIdx.x * 64 + (wv & 1) * 32;
  const int n0 = blockIdx.y * 256 + (wv >> 1) * 64;

  v8f acc[2][4] = {};
  for (int k0 = 0; k0 < K; k0 += 32) {
    v16bf a0 = load_frag_bf16(X + (size_t)m0 * K + k0, K, lane);
    v16bf a1 = load_frag_bf16(X + (size_t)(m0 + 16) * K + k0, K, lane);
#pragma unroll
    for (int j = 0; j < 4; ++j) {
      v16bf b = load_frag_bf16(W + (size_t)(n0 + j * 16) * K + k0, K, lane);
      acc[0][j] = wmma_bf16(a0, b, acc[0][j]);
      acc[1][j] = wmma_bf16(a1, b, acc[1][j]);
    }
  }
  const int cn  = lane & 15;
  const int rhi = (lane >> 4) << 3;
#pragma unroll
  for (int i = 0; i < 2; ++i)
#pragma unroll
    for (int j = 0; j < 4; ++j)
#pragma unroll
      for (int r = 0; r < 8; ++r)
        O[(size_t)(m0 + i * 16 + rhi + r) * G_ + (n0 + j * 16 + cn)] =
            acc[i][j][r];
}

// ---------------------------------------------------------------------------
// Sequential GRU scan, gate-stripe tiling. One block (WGP) per direction.
// 1024 threads = 32 waves laid out 4(M: batch rows) x 8(jg: hidden-col groups
// of 32). Wave (m0, jg) computes, for each gate g in {r,z,n}, tiles
// N = g*256 + jg*32 + jt*16 (jt=0..1) -> acc[3][2] (48-WMMA chain per step).
// r/z/n of a given (b,j) share (lane, accum-row) coords, so gates run straight
// out of the accumulators; previous hidden lives in per-thread registers.
// LDS (row pitch LP_=264 elems => conflict-free fragment loads):
//   Whh rows 0..447 (231 KB, async-preloaded) + double-buffered bf16 h (66 KB).
// Double buffering needs only ONE workgroup barrier per timestep.
// ---------------------------------------------------------------------------
__global__ __launch_bounds__(1024) void gru_scan(
    const float* __restrict__ xg0, const float* __restrict__ xg1,
    const bf16* __restrict__ Whh,   // [2][G_ x H_] row-major
    const float* __restrict__ bih,  // [2][G_]
    const float* __restrict__ bhh,  // [2][G_]
    bf16* __restrict__ Y) {         // [B_ x T_ x HH_], dir stripe of H_
  extern __shared__ char smem[];
  bf16* w_lds = (bf16*)smem;                                  // WROWS_ x LP_
  bf16* h0    = (bf16*)(smem + (size_t)WROWS_ * LP_ * 2);     // 64 x LP_
  bf16* h1    = h0 + (size_t)B_ * LP_;                        // 64 x LP_

  const int dir = blockIdx.x;
  const float* xg = dir ? xg1 : xg0;
  const bf16* Wg = Whh + (size_t)dir * G_ * H_;

  const int tid  = threadIdx.x;
  const int lane = tid & 31;
  const int wv   = tid >> 5;        // 0..31
  const int m0   = (wv & 3) * 16;   // batch-row tile
  const int jg   = wv >> 2;         // hidden cols [jg*32, jg*32+32)
  const int cn   = lane & 15;
  const int rhi  = (lane >> 4) << 3;

  // ---- async preload of Whh rows 0..WROWS_-1 into padded LDS ----
  {
    const unsigned lbase = (unsigned)(size_t)w_lds;
    const char* gsrc = (const char*)Wg;
    // 448 rows * 32 chunks of 16B
    for (int c = tid; c < WROWS_ * 32; c += 1024) {
      const int row = c >> 5, col = c & 31;
      async_ld_lds_b128(lbase + (unsigned)(row * LP_ * 2 + col * 16),
                        gsrc + (size_t)row * H_ * 2 + col * 16);
    }
  }
  // zero bf16 hidden state (buffer 0 is read at t=0)
  for (int i = tid; i < B_ * LP_ / 2; i += 1024) ((unsigned*)h0)[i] = 0u;
  wait_async0();
  __syncthreads();

  // per-thread gate biases for the 6 owned (gate, jt) columns
  const float* bi = bih + dir * G_;
  const float* bh = bhh + dir * G_;
  float bir[3][2], bhr[3][2];
#pragma unroll
  for (int g = 0; g < 3; ++g)
#pragma unroll
    for (int jt = 0; jt < 2; ++jt) {
      const int j = jg * 32 + jt * 16 + cn;
      bir[g][jt] = bi[g * H_ + j];
      bhr[g][jt] = bh[g * H_ + j];
    }

  float hreg[2][8] = {};   // persistent hidden state owned by this thread

  for (int t = 0; t < T_; ++t) {
    const bf16* hin = (t & 1) ? h1 : h0;
    bf16* hout      = (t & 1) ? h0 : h1;

    v8f acc[3][2] = {};
    for (int k0 = 0; k0 < H_; k0 += 32) {
      v16bf a = load_frag_bf16(hin + (size_t)m0 * LP_ + k0, LP_, lane);
#pragma unroll
      for (int g = 0; g < 3; ++g)
#pragma unroll
        for (int jt = 0; jt < 2; ++jt) {
          const int nrow = g * H_ + jg * 32 + jt * 16;
          v16bf b = (nrow < WROWS_)
              ? load_frag_bf16(w_lds + (size_t)nrow * LP_ + k0, LP_, lane)
              : load_frag_bf16(Wg + (size_t)nrow * H_ + k0, H_, lane);
          acc[g][jt] = wmma_bf16(a, b, acc[g][jt]);
        }
    }

#pragma unroll
    for (int jt = 0; jt < 2; ++jt) {
      const int j = jg * 32 + jt * 16 + cn;
#pragma unroll
      for (int r = 0; r < 8; ++r) {
        const int b = m0 + rhi + r;
        const float* xr = xg + ((size_t)b * T_ + t) * G_;
        const float rr =
            sigmoid_fast(xr[j] + bir[0][jt] + acc[0][jt][r] + bhr[0][jt]);
        const float zz =
            sigmoid_fast(xr[H_ + j] + bir[1][jt] + acc[1][jt][r] + bhr[1][jt]);
        const float nn =
            tanh_fast(xr[2 * H_ + j] + bir[2][jt] + rr * (acc[2][jt][r] + bhr[2][jt]));
        const float hn = (1.f - zz) * nn + zz * hreg[jt][r];
        hreg[jt][r] = hn;
        const bf16 hb = __builtin_bit_cast(bf16, f2bf_bits(hn));
        hout[b * LP_ + j] = hb;
        Y[((size_t)b * T_ + t) * HH_ + dir * H_ + j] = hb;
        if (jt == 0 && r == 0 && t + 1 < T_) {   // warm next step's xg rows
          __builtin_prefetch(xr + G_ + j, 0, 0);
          __builtin_prefetch(xr + G_ + H_ + j, 0, 0);
          __builtin_prefetch(xr + G_ + 2 * H_ + j, 0, 0);
        }
      }
    }
    __syncthreads();   // h(t) published; next step reads the other buffer
  }
}

// ---------------------------------------------------------------------------
// Gather last valid timestep + classifier: pred = hidden @ Wc^T + bc
// ---------------------------------------------------------------------------
__global__ __launch_bounds__(32) void classify(const bf16* __restrict__ Y,
                                               const int* __restrict__ seq_len,
                                               const float* __restrict__ Wc,
                                               const float* __restrict__ bc,
                                               float* __restrict__ out) {
  __shared__ float hsm[HH_];
  const int b = blockIdx.x;
  int t = seq_len[b] - 1;
  if (t < 0) t = 0;
  if (t >= T_) t = T_ - 1;
  const bf16* hp = Y + ((size_t)b * T_ + t) * HH_;
  for (int i = threadIdx.x; i < HH_; i += 32)
    hsm[i] = bf2f(__builtin_bit_cast(unsigned short, hp[i]));
  __syncthreads();
  const int c = threadIdx.x;
  float s = bc[c];
  for (int k = 0; k < HH_; ++k) s += hsm[k] * Wc[c * HH_ + k];
  out[b * 32 + c] = s;
}

// ---------------------------------------------------------------------------
extern "C" void kernel_launch(void* const* d_in, const int* in_sizes, int n_in,
                              void* d_out, int out_size, void* d_ws,
                              size_t ws_size, hipStream_t stream) {
  (void)in_sizes; (void)n_in; (void)out_size; (void)ws_size;
  const float* x     = (const float*)d_in[0];
  const int*   slen  = (const int*)d_in[1];
  const float* Wih0  = (const float*)d_in[2];
  const float* Whh0  = (const float*)d_in[3];
  const float* bih0  = (const float*)d_in[4];
  const float* bhh0  = (const float*)d_in[5];
  const float* Wih12 = (const float*)d_in[6];
  const float* Whh12 = (const float*)d_in[7];
  const float* bih12 = (const float*)d_in[8];
  const float* bhh12 = (const float*)d_in[9];
  const float* Wc    = (const float*)d_in[10];
  const float* bc    = (const float*)d_in[11];

  char* ws = (char*)d_ws;
  size_t off = 0;
  auto take = [&](size_t bytes) -> char* {
    char* p = ws + off;
    off = (off + bytes + 255) & ~(size_t)255;
    return p;
  };
  float* xg0   = (float*)take((size_t)M_ * G_ * 4);
  float* xg1   = (float*)take((size_t)M_ * G_ * 4);
  bf16* yA     = (bf16*)take((size_t)M_ * HH_ * 2);
  bf16* yB     = (bf16*)take((size_t)M_ * HH_ * 2);
  bf16* xb     = (bf16*)take((size_t)M_ * DIN_ * 2);
  bf16* wih0b  = (bf16*)take((size_t)2 * G_ * DIN_ * 2);
  bf16* whh0b  = (bf16*)take((size_t)2 * G_ * H_ * 2);
  bf16* wih12b = (bf16*)take((size_t)2 * 2 * G_ * HH_ * 2);
  bf16* whh12b = (bf16*)take((size_t)2 * 2 * G_ * H_ * 2);

  auto cvt = [&](const float* src, bf16* dst, int n) {
    cvt_bf16<<<dim3((n + 255) / 256), dim3(256), 0, stream>>>(src, dst, n);
  };
  cvt(x, xb, M_ * DIN_);
  cvt(Wih0, wih0b, 2 * G_ * DIN_);
  cvt(Whh0, whh0b, 2 * G_ * H_);
  cvt(Wih12, wih12b, 2 * 2 * G_ * HH_);
  cvt(Whh12, whh12b, 2 * 2 * G_ * H_);

  const bf16* Xin = xb;
  int K = DIN_;
  bf16* Yout = yA;
  const bf16* Yfinal = yA;
  const size_t smem_scan =
      ((size_t)WROWS_ * LP_ + 2 * (size_t)B_ * LP_) * 2;  // ~297 KB

  for (int layer = 0; layer < 3; ++layer) {
    const bf16* wih =
        layer == 0 ? wih0b : wih12b + (size_t)(layer - 1) * 2 * G_ * HH_;
    const bf16* whh =
        layer == 0 ? whh0b : whh12b + (size_t)(layer - 1) * 2 * G_ * H_;
    const float* bi = layer == 0 ? bih0 : bih12 + (size_t)(layer - 1) * 2 * G_;
    const float* bh = layer == 0 ? bhh0 : bhh12 + (size_t)(layer - 1) * 2 * G_;

    dim3 gg(M_ / 64, G_ / 256);
    xproj_gemm<<<gg, dim3(256), 0, stream>>>(Xin, wih, xg0, K);
    xproj_gemm<<<gg, dim3(256), 0, stream>>>(Xin, wih + (size_t)G_ * K, xg1, K);
    gru_scan<<<dim3(2), dim3(1024), smem_scan, stream>>>(xg0, xg1, whh, bi, bh,
                                                         Yout);
    Yfinal = Yout;
    Xin = Yout;
    K = HH_;
    Yout = (Yout == yA) ? yB : yA;
  }
  classify<<<dim3(B_), dim3(32), 0, stream>>>(Yfinal, slen, Wc, bc,
                                              (float*)d_out);
}
